// GraphTransformer_69337952026658
// MI455X (gfx1250) — matrix operand order repaired and checked by reading, not compiled, not run on hardware
//
#include <hip/hip_runtime.h>

// ---------------- problem constants (match reference) ----------------
#define NN   50000     // nodes
#define EE   600000    // edges
#define FF   128       // input features
#define ED   16        // edge_dim
#define HH   4         // heads
#define DD   32        // per-head dim
#define HID  128       // H*D
#define CC   10        // classes
#define AST  136       // padded LDS row stride (bf16 elems) for 128-wide rows
#define EST  40        // padded LDS row stride for 32-wide rows

typedef __attribute__((ext_vector_type(16))) __bf16 v16bf;
typedef __attribute__((ext_vector_type(8)))  __bf16 v8bf;
typedef __attribute__((ext_vector_type(4)))  __bf16 v4bf;
typedef __attribute__((ext_vector_type(8)))  float  v8f;

// order-preserving float->uint key for atomicMax (0 == "-inf" sentinel)
__device__ __forceinline__ unsigned fkey(float x) {
    unsigned b = __float_as_uint(x);
    return (b & 0x80000000u) ? ~b : (b | 0x80000000u);
}
__device__ __forceinline__ float funkey(unsigned k) {
    return (k & 0x80000000u) ? __uint_as_float(k ^ 0x80000000u) : __uint_as_float(~k);
}

__device__ __forceinline__ v16bf cat16(v8bf lo, v8bf hi) {
    return __builtin_shufflevector(lo, hi, 0,1,2,3,4,5,6,7,8,9,10,11,12,13,14,15);
}

// ---------------------------------------------------------------------
// K1: node projections  C = X @ W + b  for W in {Wq,Wk,Wv,Ws} (blockIdx.y)
// 256 threads / 8 waves; block computes 128 rows x 128 cols per W load.
// A staged row-major (bf16), B staged TRANSPOSED so every WMMA fragment
// is two contiguous ds_load_b128 per lane. The column-tile loop is
// unswitched on the block-uniform `full` flag and fully unrolled in the
// hot path so stores are branch-free and indices constant.
// ---------------------------------------------------------------------
__global__ __launch_bounds__(256) void gt_node_proj(
    const float* __restrict__ X,
    const float* __restrict__ Wq, const float* __restrict__ bq,
    const float* __restrict__ Wk, const float* __restrict__ bk,
    const float* __restrict__ Wv, const float* __restrict__ bv,
    const float* __restrict__ Ws, const float* __restrict__ bs,
    float* __restrict__ Q, float* __restrict__ K,
    float* __restrict__ V, float* __restrict__ S)
{
    __shared__ __bf16 As[128][AST];   // 34 KB
    __shared__ __bf16 Bt[128][AST];   // 34 KB   Bt[n][k] = W[k][n]

    const float* W; const float* bias; float* out;
    switch (blockIdx.y) {
        case 0:  W = Wq; bias = bq; out = Q; break;
        case 1:  W = Wk; bias = bk; out = K; break;
        case 2:  W = Wv; bias = bv; out = V; break;
        default: W = Ws; bias = bs; out = S; break;
    }

    const int  tid  = threadIdx.x;
    const int  m0   = blockIdx.x * 128;
    const bool full = (m0 + 128 <= NN);

    // stage A: 128 rows x 128 cols in float4 chunks (zeros past N)
    for (int ch = tid; ch < 128 * 32; ch += 256) {
        int r = ch >> 5, c = (ch & 31) << 2;
        int row = m0 + r;
        v4bf pk = {};
        if (row < NN) {
            const float4 f = *(const float4*)(X + (size_t)row * FF + c);
            pk[0] = (__bf16)f.x; pk[1] = (__bf16)f.y;
            pk[2] = (__bf16)f.z; pk[3] = (__bf16)f.w;
        }
        *(v4bf*)&As[r][c] = pk;
    }
    // stage Bt (transposed): coalesced float4 reads of W, scatter 16-bit stores
    for (int ch = tid; ch < 128 * 32; ch += 256) {
        int k = ch >> 5, c = (ch & 31) << 2;
        const float4 f = *(const float4*)(W + (size_t)k * HID + c);
        Bt[c + 0][k] = (__bf16)f.x;
        Bt[c + 1][k] = (__bf16)f.y;
        Bt[c + 2][k] = (__bf16)f.z;
        Bt[c + 3][k] = (__bf16)f.w;
    }
    __syncthreads();

    const int wave = tid >> 5, lane = tid & 31;
    const int mrow = lane & 15;
    const int aoff = (lane & 16) ? 8  : 0;   // A frag K offset (16-bit A 16x32 layout)
    const int boff = (lane & 16) ? 8  : 0;   // B frag K offset (same k order as A)
    const int arow = wave * 16 + mrow;       // this wave's row tile
    const int rbase = m0 + wave * 16 + ((lane & 16) ? 8 : 0);

    // hoist A fragments for this wave's row tile (4 K-steps)
    v16bf afrag[4];
#pragma unroll
    for (int t = 0; t < 4; ++t) {
        int kk = t * 32;
        v8bf lo = *(const v8bf*)&As[arow][kk + aoff];
        v8bf hi = *(const v8bf*)&As[arow][kk + aoff + 16];
        afrag[t] = cat16(lo, hi);
    }

    if (full) {
#pragma unroll
        for (int j = 0; j < 8; ++j) {        // 8 column tiles of 16
            v16bf bfrag[4];
#pragma unroll
            for (int t = 0; t < 4; ++t) {
                int kk = t * 32;
                v8bf blo = *(const v8bf*)&Bt[j * 16 + mrow][kk + boff];
                v8bf bhi = *(const v8bf*)&Bt[j * 16 + mrow][kk + boff + 16];
                bfrag[t] = cat16(blo, bhi);
            }
            v8f acc = {};
#pragma unroll
            for (int t = 0; t < 4; ++t)
                acc = __builtin_amdgcn_wmma_f32_16x16x32_bf16(
                    false, afrag[t], false, bfrag[t], (short)0, acc, false, false);

            int col = j * 16 + mrow;
            float bb = bias[col];
            float* op = out + (size_t)rbase * HID + col;
#pragma unroll
            for (int r = 0; r < 8; ++r) op[(size_t)r * HID] = acc[r] + bb;
        }
    } else {
        for (int j = 0; j < 8; ++j) {
            v16bf bfrag[4];
#pragma unroll
            for (int t = 0; t < 4; ++t) {
                int kk = t * 32;
                v8bf blo = *(const v8bf*)&Bt[j * 16 + mrow][kk + boff];
                v8bf bhi = *(const v8bf*)&Bt[j * 16 + mrow][kk + boff + 16];
                bfrag[t] = cat16(blo, bhi);
            }
            v8f acc = {};
#pragma unroll
            for (int t = 0; t < 4; ++t)
                acc = __builtin_amdgcn_wmma_f32_16x16x32_bf16(
                    false, afrag[t], false, bfrag[t], (short)0, acc, false, false);

            int col = j * 16 + mrow;
            float bb = bias[col];
            float* op = out + (size_t)rbase * HID + col;
#pragma unroll
            for (int r = 0; r < 8; ++r)
                if (rbase + r < NN) op[(size_t)r * HID] = acc[r] + bb;
        }
    }
}

// ---------------------------------------------------------------------
// K2: edge embedding  Eemb = edge_attr @ We  (K=16 zero-padded to 32)
// 128 edges per block, same transposed-B scheme. Output bf16.
// ---------------------------------------------------------------------
__global__ __launch_bounds__(256) void gt_edge_proj(
    const float* __restrict__ EA, const float* __restrict__ We,
    __bf16* __restrict__ Eemb)
{
    __shared__ __bf16 As[128][EST];   // 128 edges x 32 K (padded)
    __shared__ __bf16 Bt[128][EST];   // Bt[n][k], k>=16 zero

    const int  tid  = threadIdx.x;
    const int  e0   = blockIdx.x * 128;
    const bool full = (e0 + 128 <= EE);

    for (int ch = tid; ch < 128 * 8; ch += 256) {     // 4-elem chunks over 32 K
        int r = ch >> 3, c = (ch & 7) << 2;
        int e = e0 + r;
        v4bf pk = {};
        if (c < ED && e < EE) {
            const float4 f = *(const float4*)(EA + (size_t)e * ED + c);
            pk[0] = (__bf16)f.x; pk[1] = (__bf16)f.y;
            pk[2] = (__bf16)f.z; pk[3] = (__bf16)f.w;
        }
        *(v4bf*)&As[r][c] = pk;
    }
    for (int ch = tid; ch < 128 * 8; ch += 256) {     // k 0..31, c 0..127 step 4
        int k = ch >> 5, c = (ch & 31) << 2;
        if (k < ED) {
            const float4 f = *(const float4*)(We + (size_t)k * HID + c);
            Bt[c + 0][k] = (__bf16)f.x;
            Bt[c + 1][k] = (__bf16)f.y;
            Bt[c + 2][k] = (__bf16)f.z;
            Bt[c + 3][k] = (__bf16)f.w;
        } else {
            Bt[c + 0][k] = (__bf16)0.0f;
            Bt[c + 1][k] = (__bf16)0.0f;
            Bt[c + 2][k] = (__bf16)0.0f;
            Bt[c + 3][k] = (__bf16)0.0f;
        }
    }
    __syncthreads();

    const int wave = tid >> 5, lane = tid & 31;
    const int mrow = lane & 15;
    const int aoff = (lane & 16) ? 8  : 0;
    const int boff = (lane & 16) ? 8  : 0;
    const int arow = wave * 16 + mrow;
    const int rbase = e0 + wave * 16 + ((lane & 16) ? 8 : 0);

    v8bf alo = *(const v8bf*)&As[arow][aoff];
    v8bf ahi = *(const v8bf*)&As[arow][aoff + 16];
    v16bf a = cat16(alo, ahi);

    if (full) {
#pragma unroll
        for (int j = 0; j < 8; ++j) {
            v8bf blo = *(const v8bf*)&Bt[j * 16 + mrow][boff];
            v8bf bhi = *(const v8bf*)&Bt[j * 16 + mrow][boff + 16];
            v8f acc = {};
            acc = __builtin_amdgcn_wmma_f32_16x16x32_bf16(
                false, a, false, cat16(blo, bhi), (short)0, acc, false, false);
            int col = j * 16 + mrow;
            __bf16* op = Eemb + (size_t)rbase * HID + col;
#pragma unroll
            for (int r = 0; r < 8; ++r) op[(size_t)r * HID] = (__bf16)acc[r];
        }
    } else {
        for (int j = 0; j < 8; ++j) {
            v8bf blo = *(const v8bf*)&Bt[j * 16 + mrow][boff];
            v8bf bhi = *(const v8bf*)&Bt[j * 16 + mrow][boff + 16];
            v8f acc = {};
            acc = __builtin_amdgcn_wmma_f32_16x16x32_bf16(
                false, a, false, cat16(blo, bhi), (short)0, acc, false, false);
            int col = j * 16 + mrow;
            __bf16* op = Eemb + (size_t)rbase * HID + col;
#pragma unroll
            for (int r = 0; r < 8; ++r)
                if (rbase + r < EE) op[(size_t)r * HID] = (__bf16)acc[r];
        }
    }
}

// ---------------------------------------------------------------------
// K3: per-(edge,head) logit + segment max over dst (atomic key max)
// ---------------------------------------------------------------------
__global__ __launch_bounds__(256) void gt_edge_logits(
    const int* __restrict__ ei,
    const float* __restrict__ Q, const float* __restrict__ K,
    const __bf16* __restrict__ Eemb,
    float* __restrict__ logitsE, unsigned* __restrict__ maxkey)
{
    int idx = blockIdx.x * 256 + threadIdx.x;
    if (idx >= EE * HH) return;
    int e = idx >> 2, h = idx & 3;
    int src = ei[e], dst = ei[EE + e];

    const float4* q4  = (const float4*)(Q    + (size_t)dst * HID + h * DD);
    const float4* k4  = (const float4*)(K    + (size_t)src * HID + h * DD);
    const v4bf*   em4 = (const v4bf*)  (Eemb + (size_t)e   * HID + h * DD);

    float s = 0.0f;
#pragma unroll
    for (int i = 0; i < 8; ++i) {
        float4 qv = q4[i], kv = k4[i];
        v4bf   ev = em4[i];
        s += qv.x * (kv.x + (float)ev[0]) + qv.y * (kv.y + (float)ev[1])
           + qv.z * (kv.z + (float)ev[2]) + qv.w * (kv.w + (float)ev[3]);
    }
    s *= 0.17677669529663687f;  // 1/sqrt(32)

    logitsE[idx] = s;
    atomicMax(&maxkey[dst * HH + h], fkey(s));
}

// ---------------------------------------------------------------------
// K4: ex = exp(logit - max[dst]) ; denom[dst] += ex  (in place)
// ---------------------------------------------------------------------
__global__ __launch_bounds__(256) void gt_edge_exp(
    const int* __restrict__ ei,
    float* __restrict__ logitsE,
    const unsigned* __restrict__ maxkey,
    float* __restrict__ denom)
{
    int idx = blockIdx.x * 256 + threadIdx.x;
    if (idx >= EE * HH) return;
    int e = idx >> 2, h = idx & 3;
    int dst = ei[EE + e];
    float m  = funkey(maxkey[dst * HH + h]);
    float ex = __expf(logitsE[idx] - m);
    logitsE[idx] = ex;
    atomicAdd(&denom[dst * HH + h], ex);
}

// ---------------------------------------------------------------------
// K5: out[dst,c] += alpha * (V[src,c] + Eemb[e,c])  (4 channels/thread)
// ---------------------------------------------------------------------
__global__ __launch_bounds__(256) void gt_edge_scatter(
    const int* __restrict__ ei,
    const float* __restrict__ V, const __bf16* __restrict__ Eemb,
    const float* __restrict__ logitsE, const float* __restrict__ denom,
    float* __restrict__ outb)
{
    int idx = blockIdx.x * 256 + threadIdx.x;     // over EE * 32
    if (idx >= EE * 32) return;
    int e  = idx >> 5;
    int c4 = (idx & 31) << 2;
    int h  = c4 >> 5;
    int src = ei[e], dst = ei[EE + e];

    float ex    = logitsE[e * HH + h];
    float alpha = ex / (denom[dst * HH + h] + 1e-16f);

    float4 v  = *(const float4*)(V    + (size_t)src * HID + c4);
    v4bf   ev = *(const v4bf*)  (Eemb + (size_t)e   * HID + c4);
    float* ob = outb + (size_t)dst * HID + c4;
    atomicAdd(ob + 0, alpha * (v.x + (float)ev[0]));
    atomicAdd(ob + 1, alpha * (v.y + (float)ev[1]));
    atomicAdd(ob + 2, alpha * (v.z + (float)ev[2]));
    atomicAdd(ob + 3, alpha * (v.w + (float)ev[3]));
}

// ---------------------------------------------------------------------
// K6: h = relu(attn_out + skip)
// ---------------------------------------------------------------------
__global__ __launch_bounds__(256) void gt_node_finish(
    const float* __restrict__ outb, const float* __restrict__ S,
    float* __restrict__ Hout)
{
    int idx = blockIdx.x * 256 + threadIdx.x;
    if (idx >= NN * HID) return;
    float v = outb[idx] + S[idx];
    Hout[idx] = v > 0.0f ? v : 0.0f;
}

// ---------------------------------------------------------------------
// K7: classifier GEMM  logitsN = h @ Wout + bout  (10 cols padded to 16)
// 128 rows per block, single column tile.
// ---------------------------------------------------------------------
__global__ __launch_bounds__(256) void gt_out_gemm(
    const float* __restrict__ Hin, const float* __restrict__ Wout,
    const float* __restrict__ bout, float* __restrict__ logitsN)
{
    __shared__ __bf16 As[128][AST];
    __shared__ __bf16 Bt[16][AST];    // Bt[n][k]; n>=10 zero

    const int  tid  = threadIdx.x;
    const int  m0   = blockIdx.x * 128;
    const bool full = (m0 + 128 <= NN);

    for (int ch = tid; ch < 128 * 32; ch += 256) {
        int r = ch >> 5, c = (ch & 31) << 2;
        int row = m0 + r;
        v4bf pk = {};
        if (row < NN) {
            const float4 f = *(const float4*)(Hin + (size_t)row * FF + c);
            pk[0] = (__bf16)f.x; pk[1] = (__bf16)f.y;
            pk[2] = (__bf16)f.z; pk[3] = (__bf16)f.w;
        }
        *(v4bf*)&As[r][c] = pk;
    }
    for (int i = tid; i < 128 * 16; i += 256) {
        int k = i >> 4, c = i & 15;
        Bt[c][k] = (c < CC) ? (__bf16)Wout[(size_t)k * CC + c] : (__bf16)0.0f;
    }
    __syncthreads();

    const int wave = tid >> 5, lane = tid & 31;
    const int mrow = lane & 15;
    const int aoff = (lane & 16) ? 8  : 0;
    const int boff = (lane & 16) ? 8  : 0;
    const int arow = wave * 16 + mrow;
    const int rbase = m0 + wave * 16 + ((lane & 16) ? 8 : 0);

    v16bf afrag[4], bfrag[4];
#pragma unroll
    for (int t = 0; t < 4; ++t) {
        int kk = t * 32;
        v8bf alo = *(const v8bf*)&As[arow][kk + aoff];
        v8bf ahi = *(const v8bf*)&As[arow][kk + aoff + 16];
        afrag[t] = cat16(alo, ahi);
        v8bf blo = *(const v8bf*)&Bt[mrow][kk + boff];
        v8bf bhi = *(const v8bf*)&Bt[mrow][kk + boff + 16];
        bfrag[t] = cat16(blo, bhi);
    }
    v8f acc = {};
#pragma unroll
    for (int t = 0; t < 4; ++t)
        acc = __builtin_amdgcn_wmma_f32_16x16x32_bf16(
            false, afrag[t], false, bfrag[t], (short)0, acc, false, false);

    int col = mrow;
    float bb = (col < CC) ? bout[col] : 0.0f;
    float* op = logitsN + (size_t)rbase * 16 + col;
    if (full) {
#pragma unroll
        for (int r = 0; r < 8; ++r) op[r * 16] = acc[r] + bb;
    } else {
#pragma unroll
        for (int r = 0; r < 8; ++r)
            if (rbase + r < NN) op[r * 16] = acc[r] + bb;
    }
}

// ---------------------------------------------------------------------
// K8: log_softmax over 10 classes
// ---------------------------------------------------------------------
__global__ __launch_bounds__(256) void gt_logsoftmax(
    const float* __restrict__ logitsN, float* __restrict__ out)
{
    int n = blockIdx.x * 256 + threadIdx.x;
    if (n >= NN) return;
    float v[CC];
    float mx = -1e30f;
#pragma unroll
    for (int c = 0; c < CC; ++c) { v[c] = logitsN[n * 16 + c]; mx = fmaxf(mx, v[c]); }
    float sum = 0.0f;
#pragma unroll
    for (int c = 0; c < CC; ++c) sum += __expf(v[c] - mx);
    float ls = __logf(sum) + mx;
#pragma unroll
    for (int c = 0; c < CC; ++c) out[n * CC + c] = v[c] - ls;
}

// =====================================================================
extern "C" void kernel_launch(void* const* d_in, const int* in_sizes, int n_in,
                              void* d_out, int out_size, void* d_ws, size_t ws_size,
                              hipStream_t stream)
{
    // ---- inputs (setup_inputs dict order) ----
    const float* x    = (const float*)d_in[0];
    const int*   ei   = (const int*)  d_in[1];
    const float* ea   = (const float*)d_in[2];
    const float* Wq0  = (const float*)d_in[3];  const float* bq0 = (const float*)d_in[4];
    const float* Wk0  = (const float*)d_in[5];  const float* bk0 = (const float*)d_in[6];
    const float* Wv0  = (const float*)d_in[7];  const float* bv0 = (const float*)d_in[8];
    const float* We0  = (const float*)d_in[9];
    const float* Ws0  = (const float*)d_in[10]; const float* bs0 = (const float*)d_in[11];
    const float* Wq1  = (const float*)d_in[12]; const float* bq1 = (const float*)d_in[13];
    const float* Wk1  = (const float*)d_in[14]; const float* bk1 = (const float*)d_in[15];
    const float* Wv1  = (const float*)d_in[16]; const float* bv1 = (const float*)d_in[17];
    const float* We1  = (const float*)d_in[18];
    const float* Ws1  = (const float*)d_in[19]; const float* bs1 = (const float*)d_in[20];
    const float* Wout = (const float*)d_in[21]; const float* bout = (const float*)d_in[22];
    float* dout = (float*)d_out;

    // ---- workspace carve-up ----
    char* ws = (char*)d_ws;
    size_t off = 0;
    float* Qb      = (float*)(ws + off); off += (size_t)NN * HID * 4;
    float* Kb      = (float*)(ws + off); off += (size_t)NN * HID * 4;
    float* Vb      = (float*)(ws + off); off += (size_t)NN * HID * 4;
    float* Sb      = (float*)(ws + off); off += (size_t)NN * HID * 4;
    float* Hb      = (float*)(ws + off); off += (size_t)NN * HID * 4;
    float* Outb    = (float*)(ws + off); off += (size_t)NN * HID * 4;
    float* logitsE = (float*)(ws + off); off += (size_t)EE * HH * 4;
    float* denom   = (float*)(ws + off); off += (size_t)NN * HH * 4;
    float* logitsN = (float*)(ws + off); off += (size_t)NN * 16 * 4;
    unsigned* maxkey = (unsigned*)(ws + off); off += (size_t)NN * HH * 4;
    __bf16* Eemb   = (__bf16*)(ws + off);  off += (size_t)EE * HID * 2;

    const dim3 gemm_grid((NN + 127) / 128, 4);
    const int  ep_blocks = (EE + 127) / 128;
    const int  eh_blocks = (EE * HH + 255) / 256;
    const int  ec_blocks = (EE * 32 + 255) / 256;
    const int  nc_blocks = (NN * HID + 255) / 256;
    const int  og_blocks = (NN + 127) / 128;

    const float* Hin = x;
    for (int layer = 0; layer < 2; ++layer) {
        const float *Wq, *bq, *Wk, *bk, *Wv, *bv, *We, *Ws, *bs;
        if (layer == 0) { Wq=Wq0; bq=bq0; Wk=Wk0; bk=bk0; Wv=Wv0; bv=bv0; We=We0; Ws=Ws0; bs=bs0; }
        else            { Wq=Wq1; bq=bq1; Wk=Wk1; bk=bk1; Wv=Wv1; bv=bv1; We=We1; Ws=Ws1; bs=bs1; }

        hipMemsetAsync(maxkey, 0, (size_t)NN * HH * 4, stream);   // key 0 == -inf sentinel
        hipMemsetAsync(denom,  0, (size_t)NN * HH * 4, stream);
        hipMemsetAsync(Outb,   0, (size_t)NN * HID * 4, stream);

        gt_node_proj<<<gemm_grid, 256, 0, stream>>>(Hin, Wq, bq, Wk, bk, Wv, bv, Ws, bs,
                                                    Qb, Kb, Vb, Sb);
        gt_edge_proj<<<ep_blocks, 256, 0, stream>>>(ea, We, Eemb);
        gt_edge_logits<<<eh_blocks, 256, 0, stream>>>(ei, Qb, Kb, Eemb, logitsE, maxkey);
        gt_edge_exp<<<eh_blocks, 256, 0, stream>>>(ei, logitsE, maxkey, denom);
        gt_edge_scatter<<<ec_blocks, 256, 0, stream>>>(ei, Vb, Eemb, logitsE, denom, Outb);
        gt_node_finish<<<nc_blocks, 256, 0, stream>>>(Outb, Sb, Hb);
        Hin = Hb;
    }

    gt_out_gemm<<<og_blocks, 256, 0, stream>>>(Hb, Wout, bout, logitsN);
    gt_logsoftmax<<<(NN + 255) / 256, 256, 0, stream>>>(logitsN, dout);
}
